// Attention_22471268892807
// MI455X (gfx1250) — matrix operand order repaired and checked by reading, not compiled, not run on hardware
//
#include <hip/hip_runtime.h>
#include <hip/hip_bf16.h>

typedef unsigned short ushort_t;
typedef __attribute__((ext_vector_type(8)))  unsigned short ushortx8;
typedef __attribute__((ext_vector_type(8)))  __bf16        bf16x8;
typedef __attribute__((ext_vector_type(16))) __bf16        bf16x16;
typedef __attribute__((ext_vector_type(8)))  float         floatx8;

#define BATCH 4
#define CH    256
#define NTOK  4096          // 64*64
#define BN    (BATCH*NTOK)  // 16384 rows
#define QKVO  (3*CH)        // 768

#define USE_ASYNC_LDS 1

// ---------- helpers ----------

static __device__ inline ushort_t f2bf(float f) {
    unsigned u = __builtin_bit_cast(unsigned, f);
    unsigned r = u + 0x7FFFu + ((u >> 16) & 1u);   // round-to-nearest-even
    return (ushort_t)(r >> 16);
}

static __device__ inline floatx8 fzero() {
    floatx8 z;
#pragma unroll
    for (int i = 0; i < 8; ++i) z[i] = 0.0f;
    return z;
}

static __device__ inline bf16x16 combine16(ushortx8 lo, ushortx8 hi) {
    bf16x8 l = __builtin_bit_cast(bf16x8, lo);
    bf16x8 h = __builtin_bit_cast(bf16x8, hi);
    bf16x16 r;
#pragma unroll
    for (int i = 0; i < 8; ++i) { r[i] = l[i]; r[i + 8] = h[i]; }
    return r;
}

// B-fragment: 16 contiguous bf16 (lane holds column N=lane%16, K = half*16 + e).
static __device__ inline bf16x16 ld16(const ushort_t* p) {
    ushortx8 lo = *(const ushortx8*)p;
    ushortx8 hi = *(const ushortx8*)(p + 8);
    return combine16(lo, hi);
}

// A-fragment: lane (half,m) holds K = half*16 + 0..15 (elems 0..15)
// within each 32-wide K slab.
// p = &row[kslab*32 + half*16]; second half at p+8.
static __device__ inline bf16x16 ld_a(const ushort_t* p) {
    ushortx8 lo = *(const ushortx8*)p;
    ushortx8 hi = *(const ushortx8*)(p + 8);
    return combine16(lo, hi);
}

static __device__ inline floatx8 wmma_bf16(bf16x16 a, bf16x16 b, floatx8 c) {
    return __builtin_amdgcn_wmma_f32_16x16x32_bf16(
        /*neg_a=*/false, a, /*neg_b=*/false, b,
        /*c_mod=*/(short)0, c, /*reuse_a=*/false, /*reuse_b=*/false);
}

// ---------- async global -> LDS copies (CDNA5 ASYNCcnt path) ----------
#if USE_ASYNC_LDS
static __device__ inline unsigned lds_addr_of(const void* p) {
    return (unsigned)(unsigned long long)
        (__attribute__((address_space(3))) const void*)p;
}
static __device__ inline void cp_async_b128(void* lds_dst, const void* gsrc) {
    unsigned l = lds_addr_of(lds_dst);
    unsigned long long g = (unsigned long long)gsrc;
    asm volatile("global_load_async_to_lds_b128 %0, %1, off"
                 :: "v"(l), "v"(g) : "memory");
}
static __device__ inline void wait_async() {
    asm volatile("s_wait_asynccnt 0" ::: "memory");
}
#else
static __device__ inline void cp_async_b128(void* lds_dst, const void* gsrc) {
    *(ushortx8*)lds_dst = *(const ushortx8*)gsrc;
}
static __device__ inline void wait_async() {}
#endif

// ---------- kernel 0: cast weights f32 -> bf16 ----------
__global__ void cvt_weights(const float* __restrict__ wqkv,
                            const float* __restrict__ wout,
                            ushort_t* __restrict__ wqkv_bf,
                            ushort_t* __restrict__ wout_bf) {
    int i = blockIdx.x * blockDim.x + threadIdx.x;
    if (i < QKVO * CH) wqkv_bf[i] = f2bf(wqkv[i]);
    if (i < CH * CH)   wout_bf[i] = f2bf(wout[i]);
}

// ---------- kernel 1: GroupNorm, output transposed [b][n][c] bf16 ----------
__global__ __launch_bounds__(256)
void groupnorm_kernel(const float* __restrict__ x,
                      const float* __restrict__ gamma,
                      const float* __restrict__ beta,
                      ushort_t* __restrict__ xn) {
    int blk = blockIdx.x;          // b*32 + group
    int bi  = blk >> 5;
    int g   = blk & 31;
    int ch0 = g * 8;
    const float* xb = x + ((size_t)bi * CH + ch0) * NTOK;  // 8 channels x 4096

    __shared__ float red[256];
    float s = 0.f, s2 = 0.f;
    for (int i = threadIdx.x; i < 8 * NTOK; i += 256) {
        float v = xb[i];
        s += v; s2 += v * v;
    }
    red[threadIdx.x] = s;
    __syncthreads();
    for (int off = 128; off > 0; off >>= 1) {
        if ((int)threadIdx.x < off) red[threadIdx.x] += red[threadIdx.x + off];
        __syncthreads();
    }
    float tot = red[0];
    __syncthreads();
    red[threadIdx.x] = s2;
    __syncthreads();
    for (int off = 128; off > 0; off >>= 1) {
        if ((int)threadIdx.x < off) red[threadIdx.x] += red[threadIdx.x + off];
        __syncthreads();
    }
    float tot2 = red[0];

    const float inv_n = 1.0f / (8.0f * NTOK);
    float mean = tot * inv_n;
    float var  = tot2 * inv_n - mean * mean;
    float istd = rsqrtf(var + 1e-5f);

    for (int i = threadIdx.x; i < 8 * NTOK; i += 256) {
        int c  = i >> 12;          // /4096
        int sp = i & 4095;
        int ch = ch0 + c;
        float v = (xb[i] - mean) * istd * gamma[ch] + beta[ch];
        xn[((size_t)bi * NTOK + sp) * CH + ch] = f2bf(v);
    }
}

// ---------- kernel 2: QKV GEMM (16384 x 768 x 256) ----------
__global__ __launch_bounds__(256)
void qkv_gemm(const ushort_t* __restrict__ xn,
              const ushort_t* __restrict__ wq,
              const float* __restrict__ bq,
              ushort_t* __restrict__ qb,
              ushort_t* __restrict__ kb,
              ushort_t* __restrict__ vt) {
    int lane = threadIdx.x & 31, wave = threadIdx.x >> 5;
    int half = lane >> 4, lq = lane & 15;
    int row0 = blockIdx.x * 128 + wave * 16;
    int o0   = blockIdx.y * 64;

    floatx8 acc[4];
#pragma unroll
    for (int f = 0; f < 4; ++f) acc[f] = fzero();

    const ushort_t* arow = xn + (size_t)(row0 + lq) * CH;
#pragma unroll
    for (int kk = 0; kk < 8; ++kk) {
        bf16x16 a = ld_a(arow + kk * 32 + half * 16);
#pragma unroll
        for (int f = 0; f < 4; ++f) {
            const ushort_t* bp = wq + (size_t)(o0 + f * 16 + lq) * CH + kk * 32 + half * 16;
            acc[f] = wmma_bf16(a, ld16(bp), acc[f]);
        }
    }

#pragma unroll
    for (int f = 0; f < 4; ++f) {
        int o = o0 + f * 16 + lq;        // C layout: this lane's column
        float bias = bq[o];
#pragma unroll
        for (int r = 0; r < 8; ++r) {
            int gr = row0 + (half ? r + 8 : r);   // C layout: this lane's rows
            int b  = gr >> 12;
            int sp = gr & 4095;
            float v = acc[f][r] + bias;
            if (o < CH) {
                qb[((size_t)b * NTOK + sp) * CH + o] = f2bf(v * 0.0625f);
            } else if (o < 2 * CH) {
                kb[((size_t)b * NTOK + sp) * CH + (o - CH)] = f2bf(v);
            } else {
                // V stored transposed: vt[b][c][n]
                vt[((size_t)b * CH + (o - 2 * CH)) * NTOK + sp] = f2bf(v);
            }
        }
    }
}

// ---------- kernel 3: flash attention with cooperative LDS staging ----------
// Block = 4 waves, each owning 16 query rows of the same batch.
// Per 32-key step: block stages K-tile (32x256) and V-tile (256x32) into LDS
// via async b128 copies, then each wave does 16 score WMMAs + 16 PV WMMAs.
#define KT_LD 264   // 32x256 K tile, rows padded to 264 ushorts (528 B)
#define VT_LD 40    // 256x32 V tile, rows padded to 40 ushorts (80 B)

__global__ __launch_bounds__(128)
void attn_kernel(const ushort_t* __restrict__ q,
                 const ushort_t* __restrict__ k,
                 const ushort_t* __restrict__ vt,
                 ushort_t* __restrict__ ao) {
    __shared__ __align__(16) ushort_t kt[32][KT_LD];     // keys j0..j0+31, [key][c]
    __shared__ __align__(16) ushort_t vtile[256][VT_LD]; // [c][key j0..j0+31]
    __shared__ __align__(16) ushort_t ptile[4][16][40];  // per-wave P re-layout

    int tid  = threadIdx.x;
    int lane = tid & 31, wave = tid >> 5;
    int half = lane >> 4, lq = lane & 15;
    int task = blockIdx.x * 4 + wave;       // 1024 tasks = 4 batches * 256 qblocks
    int b  = task >> 8;                     // same b for all 4 waves in a block
    int i0 = (task & 255) * 16;

    const ushort_t* qbase = q  + (size_t)b * NTOK * CH;
    const ushort_t* kbase = k  + (size_t)b * NTOK * CH;
    const ushort_t* vbase = vt + (size_t)b * CH * NTOK;

    // preload Q A-fragments (16x256 as 8 slabs of K=32)
    bf16x16 aq[8];
    const ushort_t* qrow = qbase + (size_t)(i0 + lq) * CH;
#pragma unroll
    for (int kk = 0; kk < 8; ++kk) aq[kk] = ld_a(qrow + kk * 32 + half * 16);

    floatx8 oacc[16];
#pragma unroll
    for (int cc = 0; cc < 16; ++cc) oacc[cc] = fzero();

    float m_[8], l_[8];
#pragma unroll
    for (int r = 0; r < 8; ++r) { m_[r] = -1e30f; l_[r] = 0.f; }

    ushort_t (*pt)[40] = ptile[wave];

    for (int j0 = 0; j0 < NTOK; j0 += 32) {
        // ---- cooperative stage: K tile (1024 x 16B chunks), V tile (1024 x 16B) ----
#pragma unroll
        for (int it = 0; it < 8; ++it) {
            int idx  = tid + it * 128;          // 0..1023
            int krow = idx >> 5;                // 0..31
            int kc   = idx & 31;                // 16B chunk within row
            cp_async_b128(&kt[krow][kc * 8],
                          kbase + (size_t)(j0 + krow) * CH + kc * 8);
        }
#pragma unroll
        for (int it = 0; it < 8; ++it) {
            int idx  = tid + it * 128;          // 0..1023
            int vrow = idx >> 2;                // 0..255 (channel)
            int vc   = idx & 3;                 // 16B chunk within 32-key row
            cp_async_b128(&vtile[vrow][vc * 8],
                          vbase + (size_t)vrow * NTOK + j0 + vc * 8);
        }
        wait_async();
        __syncthreads();

        // ---- scores S(16x32) = Q @ K^T from LDS ----
        floatx8 s0 = fzero(), s1 = fzero();
#pragma unroll
        for (int kk = 0; kk < 8; ++kk) {
            s0 = wmma_bf16(aq[kk], ld16(&kt[lq][kk * 32 + half * 16]), s0);
            s1 = wmma_bf16(aq[kk], ld16(&kt[16 + lq][kk * 32 + half * 16]), s1);
        }

        // ---- online softmax (rows r / r+8 per half; cols across 16-lane group) ----
        float alpha[8];
#pragma unroll
        for (int r = 0; r < 8; ++r) {
            float v = fmaxf(s0[r], s1[r]);
            v = fmaxf(v, __shfl_xor(v, 1, 32));
            v = fmaxf(v, __shfl_xor(v, 2, 32));
            v = fmaxf(v, __shfl_xor(v, 4, 32));
            v = fmaxf(v, __shfl_xor(v, 8, 32));
            float mnew = fmaxf(m_[r], v);
            float p0 = __expf(s0[r] - mnew);
            float p1 = __expf(s1[r] - mnew);
            float rs = p0 + p1;
            rs += __shfl_xor(rs, 1, 32);
            rs += __shfl_xor(rs, 2, 32);
            rs += __shfl_xor(rs, 4, 32);
            rs += __shfl_xor(rs, 8, 32);
            alpha[r] = __expf(m_[r] - mnew);
            l_[r] = l_[r] * alpha[r] + rs;
            m_[r] = mnew;
            int row = half ? r + 8 : r;
            pt[row][lq]      = f2bf(p0);   // key j0+lq
            pt[row][16 + lq] = f2bf(p1);   // key j0+16+lq
        }

        // ---- rescale accumulator ----
#pragma unroll
        for (int cc = 0; cc < 16; ++cc) {
#pragma unroll
            for (int r = 0; r < 8; ++r) oacc[cc][r] *= alpha[r];
        }

        // ---- P (C-layout in LDS) -> A-fragment; O += P @ V from LDS ----
        bf16x16 pa = ld_a(&pt[lq][half * 16]);
#pragma unroll
        for (int cc = 0; cc < 16; ++cc) {
            oacc[cc] = wmma_bf16(pa, ld16(&vtile[cc * 16 + lq][half * 16]), oacc[cc]);
        }
        __syncthreads();   // protect tiles before next stage overwrites
    }

    // epilogue: normalize by l, store bf16 [b][n][c]
    ushort_t* aob = ao + (size_t)b * NTOK * CH;
#pragma unroll
    for (int cc = 0; cc < 16; ++cc) {
#pragma unroll
        for (int r = 0; r < 8; ++r) {
            int row = half ? r + 8 : r;
            float v = oacc[cc][r] / l_[r];
            aob[(size_t)(i0 + row) * CH + cc * 16 + lq] = f2bf(v);
        }
    }
}

// ---------- kernel 4: out projection + bias + residual ----------
__global__ __launch_bounds__(256)
void outproj_gemm(const ushort_t* __restrict__ ao,
                  const ushort_t* __restrict__ wo,
                  const float* __restrict__ bo,
                  const float* __restrict__ x,
                  float* __restrict__ y) {
    int lane = threadIdx.x & 31, wave = threadIdx.x >> 5;
    int half = lane >> 4, lq = lane & 15;
    int row0 = blockIdx.x * 128 + wave * 16;
    int o0   = blockIdx.y * 64;

    floatx8 acc[4];
#pragma unroll
    for (int f = 0; f < 4; ++f) acc[f] = fzero();

    const ushort_t* arow = ao + (size_t)(row0 + lq) * CH;
#pragma unroll
    for (int kk = 0; kk < 8; ++kk) {
        bf16x16 a = ld_a(arow + kk * 32 + half * 16);
#pragma unroll
        for (int f = 0; f < 4; ++f) {
            const ushort_t* bp = wo + (size_t)(o0 + f * 16 + lq) * CH + kk * 32 + half * 16;
            acc[f] = wmma_bf16(a, ld16(bp), acc[f]);
        }
    }

#pragma unroll
    for (int f = 0; f < 4; ++f) {
        int o = o0 + f * 16 + lq;
        float bias = bo[o];
#pragma unroll
        for (int r = 0; r < 8; ++r) {
            int gr = row0 + (half ? r + 8 : r);
            int b  = gr >> 12;
            int sp = gr & 4095;
            size_t idx = ((size_t)b * CH + o) * NTOK + sp;
            y[idx] = acc[f][r] + bias + x[idx];
        }
    }
}

// ---------- launch ----------
extern "C" void kernel_launch(void* const* d_in, const int* in_sizes, int n_in,
                              void* d_out, int out_size, void* d_ws, size_t ws_size,
                              hipStream_t stream) {
    const float* x      = (const float*)d_in[0];
    const float* gamma  = (const float*)d_in[1];
    const float* beta   = (const float*)d_in[2];
    const float* w_qkv  = (const float*)d_in[3];
    const float* b_qkv  = (const float*)d_in[4];
    const float* w_out  = (const float*)d_in[5];
    const float* b_out  = (const float*)d_in[6];
    float* y = (float*)d_out;

    ushort_t* ws = (ushort_t*)d_ws;
    const size_t TILE = (size_t)BN * CH;      // 16384*256
    ushort_t* xn  = ws;
    ushort_t* qb  = ws + 1 * TILE;
    ushort_t* kb  = ws + 2 * TILE;
    ushort_t* vt  = ws + 3 * TILE;
    ushort_t* ao  = ws + 4 * TILE;
    ushort_t* wqb = ws + 5 * TILE;
    ushort_t* wob = wqb + (size_t)QKVO * CH;

    cvt_weights<<<(QKVO * CH + 255) / 256, 256, 0, stream>>>(w_qkv, w_out, wqb, wob);
    groupnorm_kernel<<<BATCH * 32, 256, 0, stream>>>(x, gamma, beta, xn);
    qkv_gemm<<<dim3(BN / 128, QKVO / 64), 256, 0, stream>>>(xn, wqb, b_qkv, qb, kb, vt);
    attn_kernel<<<(BATCH * (NTOK / 16)) / 4, 128, 0, stream>>>(qb, kb, vt, ao);
    outproj_gemm<<<dim3(BN / 128, CH / 64), 256, 0, stream>>>(ao, wob, b_out, x, y);
}
